// MultiHeadAttention_16492674417330
// MI455X (gfx1250) — matrix operand order repaired and checked by reading, not compiled, not run on hardware
//
#include <hip/hip_runtime.h>
#include <hip/hip_bf16.h>

typedef __bf16 bf16;
typedef __attribute__((ext_vector_type(16))) __bf16 v16bf;
typedef __attribute__((ext_vector_type(8)))  __bf16 v8bf;
typedef __attribute__((ext_vector_type(4)))  __bf16 v4bf;
typedef __attribute__((ext_vector_type(8)))  float  v8f;

#define WMMA_BF16(a, b, c) \
  __builtin_amdgcn_wmma_f32_16x16x32_bf16(false, (a), false, (b), (short)0, (c), false, false)

// gfx1250 async global->LDS copy (ASYNCcnt path), with safe fallback
#if __has_builtin(__builtin_amdgcn_global_load_async_to_lds_b128) && \
    __has_builtin(__builtin_amdgcn_s_wait_asynccnt)
#define HAVE_ASYNC_LDS 1
#else
#define HAVE_ASYNC_LDS 0
#endif

#if HAVE_ASYNC_LDS
typedef int v4i __attribute__((vector_size(16)));
typedef __attribute__((address_space(1))) v4i glb_v4i;   // global 128-bit chunk
typedef __attribute__((address_space(3))) v4i lds_v4i;   // LDS 128-bit chunk

__device__ __forceinline__ void async_copy_b128(void* lds_ptr, const void* g_ptr) {
  // flat LDS address: low 32 bits are the LDS offset (ISA 10.2 aperture rules)
  __builtin_amdgcn_global_load_async_to_lds_b128(
      (glb_v4i*)(uintptr_t)g_ptr,
      (lds_v4i*)(unsigned)(uintptr_t)lds_ptr,
      0, 0);
}
__device__ __forceinline__ void wait_async_all() { __builtin_amdgcn_s_wait_asynccnt(0); }
#else
__device__ __forceinline__ void wait_async_all() {}
#endif

// Problem constants (match reference)
constexpr int Bq  = 4;
constexpr int Sq  = 2048;
constexpr int Dq  = 1024;
constexpr int Hq  = 16;
constexpr int NSq = 300;
constexpr int NSPq = 304;
constexpr int DGq = 19;
constexpr int DKq = 64;
constexpr int Mq  = Bq * Sq;          // 8192 rows

// ---------------------------------------------------------------------------
// fp32 -> bf16 conversion, 4 elements/thread (all our sizes are %4 == 0)
// ---------------------------------------------------------------------------
__global__ void cvt_f32_bf16_v4(const float* __restrict__ src, bf16* __restrict__ dst, int n4) {
  int i = blockIdx.x * blockDim.x + threadIdx.x;
  const int stride = gridDim.x * blockDim.x;
  for (; i < n4; i += stride) {
    const float4 v = ((const float4*)src)[i];
    v4bf o;
    o[0] = (bf16)v.x; o[1] = (bf16)v.y; o[2] = (bf16)v.z; o[3] = (bf16)v.w;
    ((v4bf*)dst)[i] = o;
  }
}

__global__ void fill_zero_bf16_v8(bf16* __restrict__ dst, int n8) {
  int i = blockIdx.x * blockDim.x + threadIdx.x;
  const int stride = gridDim.x * blockDim.x;
  const v8bf z = {};
  for (; i < n8; i += stride) ((v8bf*)dst)[i] = z;
}

// ---------------------------------------------------------------------------
// Transposing fp32 -> bf16 weight conversion: src[rows][cols] -> dst[cols][rows]
// 32x32 LDS tile, 256 threads (32 x 8)
// ---------------------------------------------------------------------------
__global__ __launch_bounds__(256)
void transpose_f32_bf16(const float* __restrict__ src, bf16* __restrict__ dst,
                        int rows, int cols) {
  __shared__ float tile[32][33];
  const int tx = threadIdx.x & 31;
  const int ty = threadIdx.x >> 5;
  const int r0 = blockIdx.y * 32;
  const int c0 = blockIdx.x * 32;
#pragma unroll
  for (int i = ty; i < 32; i += 8) {
    const int r = r0 + i, c = c0 + tx;
    tile[i][tx] = (r < rows && c < cols) ? src[(size_t)r * cols + c] : 0.0f;
  }
  __syncthreads();
#pragma unroll
  for (int i = ty; i < 32; i += 8) {
    const int c = c0 + i, r = r0 + tx;
    if (c < cols && r < rows) dst[(size_t)c * rows + r] = (bf16)tile[tx][i];
  }
}

// ---------------------------------------------------------------------------
// Fast WMMA GEMM (full tiles only): C[M,N] = A[M,K] @ Bt[N,K]^T + bias
// Requires M%64==0, N%64==0, K%32==0. 128 threads (4 waves), BM=BN=64, BK=32.
// LDS double-buffered; staging is 2x 16B contiguous copies per thread per
// operand (async global->LDS when available).
// ---------------------------------------------------------------------------
__global__ __launch_bounds__(128)
void gemm_bf16_wmma_fast(const bf16* __restrict__ A, int lda,
                         const bf16* __restrict__ Bt, int ldbt,
                         const float* __restrict__ bias,
                         void* __restrict__ Cp, int ldc,
                         int M, int N, int K, int outBf16)
{
  __shared__ bf16 As[2][64][32];   // [buf][m][k]
  __shared__ bf16 Bs[2][64][32];   // [buf][n][k]

  const int tid  = threadIdx.x;
  const int lane = tid & 31;
  const int wave = tid >> 5;
  const int l16  = lane & 15;
  const int half = lane >> 4;
  const int wm   = wave >> 1;
  const int wn   = wave & 1;
  const int m0   = blockIdx.y * 64;
  const int n0   = blockIdx.x * 64;

  // staging: thread -> (row = tid/2, khalf = (tid&1)*16), 32 bytes each operand
  const int srow = tid >> 1;
  const int skh  = (tid & 1) << 4;
  const bf16* aSrc = A  + (size_t)(m0 + srow) * lda  + skh;
  const bf16* bSrc = Bt + (size_t)(n0 + srow) * ldbt + skh;

  auto stage = [&](int buf, int k0) {
#if HAVE_ASYNC_LDS
    async_copy_b128(&As[buf][srow][skh],     aSrc + k0);
    async_copy_b128(&As[buf][srow][skh + 8], aSrc + k0 + 8);
    async_copy_b128(&Bs[buf][srow][skh],     bSrc + k0);
    async_copy_b128(&Bs[buf][srow][skh + 8], bSrc + k0 + 8);
#else
    __builtin_memcpy(&As[buf][srow][skh], aSrc + k0, 32);
    __builtin_memcpy(&Bs[buf][srow][skh], bSrc + k0, 32);
#endif
  };

  v8f acc[2][2] = {};
  const int KT = K >> 5;

  stage(0, 0);
  for (int kt = 0; kt < KT; ++kt) {
    const int cur = kt & 1;
    wait_async_all();
    __syncthreads();
    if (kt + 1 < KT) stage(cur ^ 1, (kt + 1) << 5);   // prefetch next slab

    v16bf fa[2], fb[2];
#pragma unroll
    for (int ti = 0; ti < 2; ++ti) {
      const int r = wm * 32 + ti * 16 + l16;
#pragma unroll
      for (int e = 0; e < 8; ++e) {
        fa[ti][e]     = As[cur][r][half * 16 + e];
        fa[ti][8 + e] = As[cur][r][half * 16 + 8 + e];
      }
    }
#pragma unroll
    for (int tj = 0; tj < 2; ++tj) {
      const int c = wn * 32 + tj * 16 + l16;
      fb[tj] = *(const v16bf*)&Bs[cur][c][half * 16];
    }
#pragma unroll
    for (int ti = 0; ti < 2; ++ti)
#pragma unroll
      for (int tj = 0; tj < 2; ++tj)
        acc[ti][tj] = WMMA_BF16(fa[ti], fb[tj], acc[ti][tj]);
  }

  // epilogue (no guards: full tiles)
#pragma unroll
  for (int ti = 0; ti < 2; ++ti)
#pragma unroll
    for (int tj = 0; tj < 2; ++tj)
#pragma unroll
      for (int r = 0; r < 8; ++r) {
        const int grow = m0 + wm * 32 + ti * 16 + r + 8 * half;
        const int gcol = n0 + wn * 32 + tj * 16 + l16;
        float v = acc[ti][tj][r] + bias[gcol];
        if (outBf16) ((bf16*)Cp)[(size_t)grow * ldc + gcol] = (bf16)v;
        else         ((float*)Cp)[(size_t)grow * ldc + gcol] = v;
      }
}

// ---------------------------------------------------------------------------
// Edge WMMA GEMM: same math, hoisted guards (used for K=300 / N=300 cases).
// ---------------------------------------------------------------------------
__global__ __launch_bounds__(128)
void gemm_bf16_wmma_edge(const bf16* __restrict__ A, int lda,
                         const bf16* __restrict__ Bt, int ldbt,
                         const float* __restrict__ bias,
                         void* __restrict__ Cp, int ldc,
                         int M, int N, int K, int outBf16)
{
  __shared__ bf16 As[64][32];
  __shared__ bf16 Bs[64][32];

  const int tid  = threadIdx.x;
  const int lane = tid & 31;
  const int wave = tid >> 5;
  const int l16  = lane & 15;
  const int half = lane >> 4;
  const int wm   = wave >> 1;
  const int wn   = wave & 1;
  const int m0   = blockIdx.y * 64;
  const int n0   = blockIdx.x * 64;

  const int srow = tid >> 1;
  const int skh  = (tid & 1) << 4;
  const int agrow = m0 + srow;
  const int bgrow = n0 + srow;
  const bf16* aSrc = A  + (size_t)agrow * lda  + skh;
  const bf16* bSrc = Bt + (size_t)bgrow * ldbt + skh;

  v8f acc[2][2] = {};
  const int KT = (K + 31) >> 5;

  for (int kt = 0; kt < KT; ++kt) {
    const int k0 = kt << 5;
    __syncthreads();
    {
      bf16* d = &As[srow][skh];
      if (agrow < M && (k0 + skh + 16) <= K) {
        __builtin_memcpy(d, aSrc + k0, 32);
      } else {
        int lim = K - (k0 + skh); if (agrow >= M) lim = 0;
#pragma unroll
        for (int e = 0; e < 16; ++e) d[e] = (e < lim) ? aSrc[k0 + e] : (bf16)0.0f;
      }
    }
    {
      bf16* d = &Bs[srow][skh];
      if (bgrow < N && (k0 + skh + 16) <= K) {
        __builtin_memcpy(d, bSrc + k0, 32);
      } else {
        int lim = K - (k0 + skh); if (bgrow >= N) lim = 0;
#pragma unroll
        for (int e = 0; e < 16; ++e) d[e] = (e < lim) ? bSrc[k0 + e] : (bf16)0.0f;
      }
    }
    __syncthreads();

    v16bf fa[2], fb[2];
#pragma unroll
    for (int ti = 0; ti < 2; ++ti) {
      const int r = wm * 32 + ti * 16 + l16;
#pragma unroll
      for (int e = 0; e < 8; ++e) {
        fa[ti][e]     = As[r][half * 16 + e];
        fa[ti][8 + e] = As[r][half * 16 + 8 + e];
      }
    }
#pragma unroll
    for (int tj = 0; tj < 2; ++tj) {
      const int c = wn * 32 + tj * 16 + l16;
      fb[tj] = *(const v16bf*)&Bs[c][half * 16];
    }
#pragma unroll
    for (int ti = 0; ti < 2; ++ti)
#pragma unroll
      for (int tj = 0; tj < 2; ++tj)
        acc[ti][tj] = WMMA_BF16(fa[ti], fb[tj], acc[ti][tj]);
  }

#pragma unroll
  for (int ti = 0; ti < 2; ++ti)
#pragma unroll
    for (int tj = 0; tj < 2; ++tj)
#pragma unroll
      for (int r = 0; r < 8; ++r) {
        const int grow = m0 + wm * 32 + ti * 16 + r + 8 * half;
        const int gcol = n0 + wn * 32 + tj * 16 + l16;
        if (grow < M && gcol < N) {
          const float v = acc[ti][tj][r] + bias[gcol];
          if (outBf16) ((bf16*)Cp)[(size_t)grow * ldc + gcol] = (bf16)v;
          else         ((float*)Cp)[(size_t)grow * ldc + gcol] = v;
        }
      }
}

// ---------------------------------------------------------------------------
// Flash-style causal attention, WMMA on both matmuls. One wave per block.
// Q,K: bf16 [B*S, D]; Vp: bf16 [B*S, NSP] zero-padded; agg: bf16 [B*S, NSP].
// ---------------------------------------------------------------------------
__global__ __launch_bounds__(32)
void attn_wmma(const bf16* __restrict__ Q, const bf16* __restrict__ Km,
               const bf16* __restrict__ Vp, bf16* __restrict__ agg)
{
  __shared__ bf16 Plds[16][32];

  const int lane = threadIdx.x;
  const int l16  = lane & 15;
  const int half = lane >> 4;
  const int q0   = blockIdx.x * 16;
  const int b    = blockIdx.y / Hq;
  const int h    = blockIdx.y % Hq;

  v16bf aQ[2];
  {
    const bf16* qb = Q + (size_t)(b * Sq + q0 + l16) * Dq + h * DKq;
#pragma unroll
    for (int dc = 0; dc < 2; ++dc) {
      v8bf lo = *(const v8bf*)(qb + dc * 32 + half * 16);
      v8bf hi = *(const v8bf*)(qb + dc * 32 + half * 16 + 8);
#pragma unroll
      for (int e = 0; e < 8; ++e) { aQ[dc][e] = lo[e]; aQ[dc][8 + e] = hi[e]; }
    }
  }

  v8f o0 = {}, o1 = {};
  float mrow[8], lrow[8];
#pragma unroll
  for (int r = 0; r < 8; ++r) { mrow[r] = -INFINITY; lrow[r] = 0.0f; }

  const int nsteps = (q0 + 16 + 31) >> 5;
  for (int j = 0; j < nsteps; ++j) {
    const int k0 = j << 5;

    v8f sc0 = {}, sc1 = {};
    {
      const bf16* kb0 = Km + (size_t)(b * Sq + k0 + l16) * Dq + h * DKq;
      const bf16* kb1 = Km + (size_t)(b * Sq + k0 + 16 + l16) * Dq + h * DKq;
#pragma unroll
      for (int dc = 0; dc < 2; ++dc) {
        v16bf b0 = *(const v16bf*)(kb0 + dc * 32 + half * 16);
        v16bf b1 = *(const v16bf*)(kb1 + dc * 32 + half * 16);
        sc0 = WMMA_BF16(aQ[dc], b0, sc0);
        sc1 = WMMA_BF16(aQ[dc], b1, sc1);
      }
    }

    __syncthreads();
#pragma unroll
    for (int r = 0; r < 8; ++r) {
      const int q = q0 + r + 8 * half;
      float s0 = sc0[r] * 0.125f; if (k0 + l16 > q)      s0 = -1e9f;
      float s1 = sc1[r] * 0.125f; if (k0 + 16 + l16 > q) s1 = -1e9f;
      float mx = fmaxf(s0, s1);
#pragma unroll
      for (int off = 1; off < 16; off <<= 1) mx = fmaxf(mx, __shfl_xor(mx, off, 16));
      const float mn   = fmaxf(mrow[r], mx);
      const float corr = __expf(mrow[r] - mn);
      const float p0 = __expf(s0 - mn);
      const float p1 = __expf(s1 - mn);
      float ps = p0 + p1;
#pragma unroll
      for (int off = 1; off < 16; off <<= 1) ps += __shfl_xor(ps, off, 16);
      lrow[r] = lrow[r] * corr + ps;
      mrow[r] = mn;
      o0[r] *= corr; o1[r] *= corr;
      Plds[r + 8 * half][l16]      = (bf16)p0;
      Plds[r + 8 * half][16 + l16] = (bf16)p1;
    }
    __syncthreads();

    v16bf aP;
#pragma unroll
    for (int e = 0; e < 8; ++e) {
      aP[e]     = Plds[l16][half * 16 + e];
      aP[8 + e] = Plds[l16][half * 16 + 8 + e];
    }

    v16bf bv0, bv1;
    {
      const bf16* vbase = Vp + (size_t)(b * Sq + k0 + half * 16) * NSPq + h * DGq;
#pragma unroll
      for (int e = 0; e < 16; ++e) bv0[e] = vbase[(size_t)e * NSPq + l16];
      if (16 + l16 < DGq) {
#pragma unroll
        for (int e = 0; e < 16; ++e) bv1[e] = vbase[(size_t)e * NSPq + 16 + l16];
      } else {
#pragma unroll
        for (int e = 0; e < 16; ++e) bv1[e] = (bf16)0.0f;
      }
    }
    o0 = WMMA_BF16(aP, bv0, o0);
    o1 = WMMA_BF16(aP, bv1, o1);
  }

#pragma unroll
  for (int r = 0; r < 8; ++r) {
    const int q = q0 + r + 8 * half;
    const float inv = 1.0f / lrow[r];
    bf16* arow = agg + (size_t)(b * Sq + q) * NSPq + h * DGq;
    arow[l16] = (bf16)(o0[r] * inv);                               // l16 < 16 <= DG
    if (16 + l16 < DGq) arow[16 + l16] = (bf16)(o1[r] * inv);
  }
}

// ---------------------------------------------------------------------------
// Host launcher
// ---------------------------------------------------------------------------
extern "C" void kernel_launch(void* const* d_in, const int* in_sizes, int n_in,
                              void* d_out, int out_size, void* d_ws, size_t ws_size,
                              hipStream_t stream)
{
  (void)in_sizes; (void)n_in; (void)out_size; (void)ws_size;
  const float* ctx = (const float*)d_in[0];
  const float* val = (const float*)d_in[1];
  // d_in[2] = causal mask (triu bool) — computed analytically in-kernel
  const float* Wq = (const float*)d_in[3];  const float* bq = (const float*)d_in[4];
  const float* Wk = (const float*)d_in[5];  const float* bk = (const float*)d_in[6];
  const float* Wv = (const float*)d_in[7];  const float* bv = (const float*)d_in[8];
  const float* Wg = (const float*)d_in[9];  const float* bg = (const float*)d_in[10];
  const float* Wo = (const float*)d_in[11]; const float* bo = (const float*)d_in[12];

  char* ws = (char*)d_ws;
  size_t off = 0;
  auto alloc = [&](size_t bytes) -> void* {
    void* p = ws + off;
    off += (bytes + 255) & ~(size_t)255;
    return p;
  };
  bf16* Xb   = (bf16*)alloc((size_t)Mq * Dq * 2);      // context bf16
  bf16* Vb   = (bf16*)alloc((size_t)Mq * NSq * 2);     // value bf16
  bf16* Wqt  = (bf16*)alloc((size_t)Dq * Dq * 2);      // transposed weights [out][in]
  bf16* Wkt  = (bf16*)alloc((size_t)Dq * Dq * 2);
  bf16* Wvt  = (bf16*)alloc((size_t)NSq * NSq * 2);
  bf16* Wgt  = (bf16*)alloc((size_t)Dq * NSq * 2);
  bf16* Wot  = (bf16*)alloc((size_t)Dq * Dq * 2);
  bf16* Qb   = (bf16*)alloc((size_t)Mq * Dq * 2);
  bf16* Kb   = (bf16*)alloc((size_t)Mq * Dq * 2);
  bf16* Vpb  = (bf16*)alloc((size_t)Mq * NSPq * 2);    // padded V projection
  bf16* Aggb = (bf16*)alloc((size_t)Mq * NSPq * 2);    // attention gains
  bf16* O1b  = (bf16*)alloc((size_t)Mq * Dq * 2);      // after gain_to_context

  // activations: straight conversion
  {
    int n4 = (Mq * Dq) / 4;
    cvt_f32_bf16_v4<<<4096, 256, 0, stream>>>(ctx, Xb, n4);
    n4 = (Mq * NSq) / 4;
    cvt_f32_bf16_v4<<<4096, 256, 0, stream>>>(val, Vb, n4);
  }
  // weights: convert + transpose  src[K,N] -> dst[N,K]
  auto wtr = [&](const float* s, bf16* d, int rows, int cols) {
    dim3 g((cols + 31) / 32, (rows + 31) / 32);
    transpose_f32_bf16<<<g, 256, 0, stream>>>(s, d, rows, cols);
  };
  wtr(Wq, Wqt, Dq, Dq);
  wtr(Wk, Wkt, Dq, Dq);
  wtr(Wv, Wvt, NSq, NSq);
  wtr(Wg, Wgt, NSq, Dq);
  wtr(Wo, Wot, Dq, Dq);

  // zero-fill padded V projection buffer (cols NS..NSP-1 must be 0)
  fill_zero_bf16_v8<<<2048, 256, 0, stream>>>(Vpb, (Mq * NSPq) / 8);

  const dim3 blk(128);
  auto grid = [](int M, int N) { return dim3((unsigned)((N + 63) / 64), (unsigned)((M + 63) / 64)); };

  // Q = X @ Wq + bq ; K = X @ Wk + bk   (fast path: 8192x1024x1024)
  gemm_bf16_wmma_fast<<<grid(Mq, Dq), blk, 0, stream>>>(Xb, Dq, Wqt, Dq, bq, Qb, Dq, Mq, Dq, Dq, 1);
  gemm_bf16_wmma_fast<<<grid(Mq, Dq), blk, 0, stream>>>(Xb, Dq, Wkt, Dq, bk, Kb, Dq, Mq, Dq, Dq, 1);
  // V_proj = value @ Wv + bv -> [B*S, NSP] (edge path: N=K=300)
  gemm_bf16_wmma_edge<<<grid(Mq, NSq), blk, 0, stream>>>(Vb, NSq, Wvt, NSq, bv, Vpb, NSPq, Mq, NSq, NSq, 1);

  // attention: agg[B*S, NSP] (bf16)
  attn_wmma<<<dim3(Sq / 16, Bq * Hq), 32, 0, stream>>>(Qb, Kb, Vpb, Aggb);

  // out1 = agg[:, :NS] @ Wg + bg   (edge path: K=300)
  gemm_bf16_wmma_edge<<<grid(Mq, Dq), blk, 0, stream>>>(Aggb, NSPq, Wgt, NSq, bg, O1b, Dq, Mq, Dq, NSq, 1);
  // out  = out1 @ Wo + bo          (fast path, fp32 out -> d_out)
  gemm_bf16_wmma_fast<<<grid(Mq, Dq), blk, 0, stream>>>(O1b, Dq, Wot, Dq, bo, d_out, Dq, Mq, Dq, Dq, 0);
}
